// ImportanceWeightedMoE_71854802862233
// MI455X (gfx1250) — matrix operand order repaired and checked by reading, not compiled, not run on hardware
//
#include <hip/hip_runtime.h>
#include <hip/hip_bf16.h>
#include <math.h>

// ---- problem constants ----
#define E_    16
#define DH_   1024
#define DC_   512
#define H2_   2048
#define OUT_  64
#define KTOT  1536      // DH + DC
#define NTOK  4096      // B*T
#define TT    512       // T
#define BB    8         // B
#define TAU_  0.1f

typedef __attribute__((ext_vector_type(16))) __bf16          v16bf;
typedef __attribute__((ext_vector_type(2)))  __bf16          v2bf;
typedef __attribute__((ext_vector_type(8)))  float           v8f;
typedef __attribute__((ext_vector_type(8)))  unsigned short  u16x8;

// native converts (RNE) -> compiler picks v_cvt_pk_bf16_f32 class ops
static __device__ __forceinline__ unsigned pack2bf(float a, float b) {
  union { v2bf v; unsigned u; } p;
  p.v[0] = (__bf16)a;
  p.v[1] = (__bf16)b;
  return p.u;
}

static __device__ __forceinline__ float wave_sum(float v) {
#pragma unroll
  for (int m = 16; m >= 1; m >>= 1) v += __shfl_xor(v, m, 32);
  return v;
}
static __device__ __forceinline__ float wave_max(float v) {
#pragma unroll
  for (int m = 16; m >= 1; m >>= 1) v = fmaxf(v, __shfl_xor(v, m, 32));
  return v;
}

static __device__ __forceinline__ float gelu_exact(float x) {
  return 0.5f * x * (1.0f + erff(x * 0.70710678118654752440f));
}

// A fragment: 16x32 bf16, lane L holds row M=L%16; vector elems 0..15 = K
// kb+16*half .. +15 (contiguous, matches B)
static __device__ __forceinline__ v16bf load_A_lds(const __bf16* row, int kb, int hf) {
  union { v16bf v; u16x8 u[2]; } f;
  f.u[0] = *(const u16x8*)(row + kb + 16 * hf);
  f.u[1] = *(const u16x8*)(row + kb + 16 * hf + 8);
  return f.v;
}

// B fragment: 32x16 bf16, lane L holds column N=L%16, K = 16 contiguous fp32
// (pointer already includes the +16*half term). Convert on the fly.
static __device__ __forceinline__ v16bf load_B_g(const float* p) {
  union { v16bf v; unsigned w[8]; } f;
  const float4* q = (const float4*)p;
#pragma unroll
  for (int j = 0; j < 4; ++j) {
    float4 t = q[j];
    f.w[2 * j + 0] = pack2bf(t.x, t.y);
    f.w[2 * j + 1] = pack2bf(t.z, t.w);
  }
  return f.v;
}

static __device__ __forceinline__ v8f wmma_bf16(v16bf a, v16bf b, v8f c) {
  return __builtin_amdgcn_wmma_f32_16x16x32_bf16(false, a, false, b, (short)0, c,
                                                 false, false);
}

// ---------------------------------------------------------------------------
// 0) zero output + expert counters
__global__ void k_zero(float* out, int n, int* cnt) {
  int i = blockIdx.x * blockDim.x + threadIdx.x;
  if (i < E_) cnt[i] = 0;
  for (int j = i; j < n; j += gridDim.x * blockDim.x) out[j] = 0.f;
}

// 1) normalized anchors
__global__ void k_norm_anchor(const float* __restrict__ anchor, float* __restrict__ na) {
  int e = blockIdx.x, tid = threadIdx.x, lane = tid & 31, wv = tid >> 5;
  __shared__ float row[DC_];
  __shared__ float wred[8];
  __shared__ float sinv;
  for (int i = tid; i < DC_; i += 256) row[i] = anchor[e * DC_ + i];
  __syncthreads();
  float ss = 0.f;
  for (int i = tid; i < DC_; i += 256) ss += row[i] * row[i];
  ss = wave_sum(ss);
  if (lane == 0) wred[wv] = ss;
  __syncthreads();
  if (tid == 0) {
    float s = 0.f;
    for (int w = 0; w < 8; ++w) s += wred[w];
    sinv = 1.0f / fmaxf(sqrtf(s), 1e-12f);
  }
  __syncthreads();
  for (int i = tid; i < DC_; i += 256) na[e * DC_ + i] = row[i] * sinv;
}

// 2) imp = softmax(fi / clamp(temp)) per expert row + entropy
__global__ void k_imp(const float* __restrict__ fi, const float* __restrict__ temperature,
                      float* __restrict__ imp, float* __restrict__ ent_imp) {
  int e = blockIdx.x, tid = threadIdx.x, lane = tid & 31, wv = tid >> 5;
  __shared__ float row[DH_];
  __shared__ float wred[8];
  __shared__ float bval;
  float temp = fminf(fmaxf(temperature[0], 0.1f), 5.0f);
  for (int i = tid; i < DH_; i += 256) row[i] = fi[e * DH_ + i] / temp;
  __syncthreads();
  float mx = -1e30f;
  for (int i = tid; i < DH_; i += 256) mx = fmaxf(mx, row[i]);
  mx = wave_max(mx);
  if (lane == 0) wred[wv] = mx;
  __syncthreads();
  if (tid == 0) {
    float m = wred[0];
    for (int w = 1; w < 8; ++w) m = fmaxf(m, wred[w]);
    bval = m;
  }
  __syncthreads();
  float M = bval;
  __syncthreads();
  float s = 0.f;
  for (int i = tid; i < DH_; i += 256) {
    float x = expf(row[i] - M);
    row[i] = x;
    s += x;
  }
  s = wave_sum(s);
  __syncthreads();
  if (lane == 0) wred[wv] = s;
  __syncthreads();
  if (tid == 0) {
    float a = 0.f;
    for (int w = 0; w < 8; ++w) a += wred[w];
    bval = a;
  }
  __syncthreads();
  float inv = 1.0f / bval;
  float hsum = 0.f;
  for (int i = tid; i < DH_; i += 256) {
    float p = row[i] * inv;
    imp[e * DH_ + i] = p;
    hsum -= p * logf(p + 1e-8f);
  }
  hsum = wave_sum(hsum);
  __syncthreads();
  if (lane == 0) wred[wv] = hsum;
  __syncthreads();
  if (tid == 0) {
    float a = 0.f;
    for (int w = 0; w < 8; ++w) a += wred[w];
    ent_imp[e] = a;
  }
}

// 3) router: cosine logits + gumbel softmax, top-2 mask, expert token lists
__global__ void k_router(const float* __restrict__ ce, const float* __restrict__ na,
                         const float* __restrict__ gumbel, float* __restrict__ ew,
                         int* __restrict__ cnt, int* __restrict__ toklist) {
  int t = blockIdx.x;
  int tid = threadIdx.x, lane = tid & 31, wv = tid >> 5;
  __shared__ float sce[DC_];
  __shared__ float wred[8];
  __shared__ float sinv;
  __shared__ float logits[E_];
  for (int i = tid; i < DC_; i += 256) sce[i] = ce[(size_t)t * DC_ + i];
  __syncthreads();
  float ss = 0.f;
  for (int i = tid; i < DC_; i += 256) ss += sce[i] * sce[i];
  ss = wave_sum(ss);
  if (lane == 0) wred[wv] = ss;
  __syncthreads();
  if (tid == 0) {
    float s = 0.f;
    for (int w = 0; w < 8; ++w) s += wred[w];
    sinv = 0.125f / fmaxf(sqrtf(s), 1e-12f);  // fold the 0.125 scale in
  }
  __syncthreads();
  for (int a = wv; a < E_; a += 8) {
    float d = 0.f;
    for (int i = lane; i < DC_; i += 32) d += sce[i] * na[a * DC_ + i];
    d = wave_sum(d);
    if (lane == 0) logits[a] = d * sinv;
  }
  __syncthreads();
  if (tid == 0) {
    float l[E_], mx = -1e30f;
    for (int e = 0; e < E_; ++e) {
      l[e] = (logits[e] + gumbel[(size_t)t * E_ + e]) * (1.0f / TAU_);
      mx = fmaxf(mx, l[e]);
    }
    float s = 0.f;
    for (int e = 0; e < E_; ++e) { l[e] = expf(l[e] - mx); s += l[e]; }
    float inv = 1.0f / s;
    for (int e = 0; e < E_; ++e) l[e] *= inv;
    int i1 = 0; float v1 = -1.f;
    for (int e = 0; e < E_; ++e) if (l[e] > v1) { v1 = l[e]; i1 = e; }   // first max (top_k tie rule)
    int i2 = -1; float v2 = -1.f;
    for (int e = 0; e < E_; ++e) if (e != i1 && l[e] > v2) { v2 = l[e]; i2 = e; }
    for (int e = 0; e < E_; ++e)
      ew[(size_t)t * E_ + e] = (e == i1 || e == i2) ? l[e] : 0.f;
    int p1 = atomicAdd(cnt + i1, 1); toklist[i1 * NTOK + p1] = t;
    int p2 = atomicAdd(cnt + i2, 1); toklist[i2 * NTOK + p2] = t;
  }
}

// 4) counts[tt,e] = sum_b ew[b,tt,e]
__global__ void k_counts(const float* __restrict__ ew, float* __restrict__ counts) {
  int i = blockIdx.x * blockDim.x + threadIdx.x;
  if (i < TT * E_) {
    int tt = i / E_, e = i - tt * E_;
    float s = 0.f;
    for (int b = 0; b < BB; ++b) s += ew[((size_t)(b * TT + tt)) * E_ + e];
    counts[i] = s;
  }
}

// 5) aux loss = 0.5*(std(counts,ddof=1) + H(load)) - 0.01*mean(H(imp))
__global__ void k_loss(const float* __restrict__ counts, const float* __restrict__ ent_imp,
                       float* __restrict__ out_aux) {
  int tid = threadIdx.x, lane = tid & 31, wv = tid >> 5;
  const int n = TT * E_;
  __shared__ float r1[8], r2[8];
  __shared__ float Ssum, Ssq;
  float s = 0.f, s2 = 0.f;
  for (int i = tid; i < n; i += 256) {
    float c = counts[i];
    s += c; s2 += c * c;
  }
  s = wave_sum(s); s2 = wave_sum(s2);
  if (lane == 0) { r1[wv] = s; r2[wv] = s2; }
  __syncthreads();
  if (tid == 0) {
    float a = 0.f, b = 0.f;
    for (int w = 0; w < 8; ++w) { a += r1[w]; b += r2[w]; }
    Ssum = a; Ssq = b;
  }
  __syncthreads();
  float S = Ssum + 1e-8f;
  float h = 0.f;
  for (int i = tid; i < n; i += 256) {
    float load = counts[i] / S;
    h -= load * logf(load + 1e-8f);
  }
  h = wave_sum(h);
  __syncthreads();
  if (lane == 0) r1[wv] = h;
  __syncthreads();
  if (tid == 0) {
    float H = 0.f;
    for (int w = 0; w < 8; ++w) H += r1[w];
    float var = (Ssq - Ssum * Ssum / (float)n) / (float)(n - 1);
    float sd = sqrtf(fmaxf(var, 0.f));
    float me = 0.f;
    for (int e = 0; e < E_; ++e) me += ent_imp[e];
    me *= (1.0f / 16.0f);
    out_aux[0] = 0.5f * (sd + H) - 0.01f * me;
  }
}

// ---------------------------------------------------------------------------
// 6) sparse expert MLP: one block = (expert, 16-token tile), 8 waves, bf16 WMMA
#define XSTRIDE 1544   // bf16 elems per X row (1536 + 8 pad; keeps 16B align)
#define ASTRIDE 2056   // bf16 elems per act row (2048 + 8 pad)
#define SMEM_X_BYTES   (16 * XSTRIDE * 2)   // 49408, reused for L2 partials
#define SMEM_ACT_BYTES (16 * ASTRIDE * 2)   // 65792

__global__ __launch_bounds__(256) void k_moe(
    const float* __restrict__ h, const float* __restrict__ ce,
    const float* __restrict__ imp, const float* __restrict__ W1h,
    const float* __restrict__ W1c, const float* __restrict__ b1,
    const float* __restrict__ W2, const float* __restrict__ b2,
    const float* __restrict__ ew, const int* __restrict__ cnt,
    const int* __restrict__ toklist, float* __restrict__ out) {
  __shared__ char smem[SMEM_X_BYTES + SMEM_ACT_BYTES];
  __shared__ int toks[16];
  __bf16* X = (__bf16*)smem;                       // [16][XSTRIDE] bf16
  __bf16* ACT = (__bf16*)(smem + SMEM_X_BYTES);    // [16][ASTRIDE] bf16
  float* partial = (float*)smem;                   // [8 waves][16*64] f32 (aliases X)

  int e = blockIdx.y;
  int tileBase = blockIdx.x * 16;
  int m_e = cnt[e];
  if (tileBase >= m_e) return;  // uniform across block

  int tid = threadIdx.x, lane = tid & 31, wv = tid >> 5;
  int hf = lane >> 4;    // half-wave select
  int lm = lane & 15;

  if (tid < 16) {
    int idx = tileBase + tid;
    toks[tid] = (idx < m_e) ? toklist[e * NTOK + idx] : -1;
  }
  __syncthreads();

  // stage X = [h*imp[e] | code_emb] as bf16 pairs (zeros for padded rows)
  {
    unsigned* X32 = (unsigned*)X;  // row stride XSTRIDE/2 dwords
    for (int i = tid; i < 16 * (KTOT / 2); i += 256) {
      int m = i / (KTOT / 2), dd = (i - m * (KTOT / 2)) * 2;
      int t = toks[m];
      float a = 0.f, b = 0.f;
      if (t >= 0) {
        if (dd < DH_) {
          a = h[(size_t)t * DH_ + dd]     * imp[e * DH_ + dd];
          b = h[(size_t)t * DH_ + dd + 1] * imp[e * DH_ + dd + 1];
        } else {
          a = ce[(size_t)t * DC_ + (dd - DH_)];
          b = ce[(size_t)t * DC_ + (dd - DH_) + 1];
        }
      }
      X32[m * (XSTRIDE / 2) + (dd >> 1)] = pack2bf(a, b);
    }
  }
  __syncthreads();

  // ---- layer 1: hidden[16 x 2048] = X[16 x 1536] @ W1cat^T, +b1, GELU ----
  // Each wave owns 16 H2-column tiles, processed in 4 groups of 4 so one
  // A fragment (LDS) feeds 4 independent WMMAs.
  const __bf16* Xrow = X + lm * XSTRIDE;
  for (int g = 0; g < 4; ++g) {
    int ncol[4];
    const float* rowh[4];
    const float* rowc[4];
#pragma unroll
    for (int j = 0; j < 4; ++j) {
      int n = (wv * 16 + g * 4 + j) * 16 + lm;
      ncol[j] = n;
      rowh[j] = W1h + ((size_t)e * H2_ + n) * DH_ + 16 * hf;
      rowc[j] = W1c + ((size_t)e * H2_ + n) * DC_ + 16 * hf;
    }
    v8f acc[4] = {};
#pragma unroll 2
    for (int ks = 0; ks < DH_ / 32; ++ks) {   // W1h region, no branch
      int kb = ks * 32;
      v16bf a = load_A_lds(Xrow, kb, hf);
#pragma unroll
      for (int j = 0; j < 4; ++j)
        acc[j] = wmma_bf16(a, load_B_g(rowh[j] + kb), acc[j]);
    }
#pragma unroll 2
    for (int ks = 0; ks < DC_ / 32; ++ks) {   // W1c region, no branch
      int kb = ks * 32;
      v16bf a = load_A_lds(Xrow, DH_ + kb, hf);
#pragma unroll
      for (int j = 0; j < 4; ++j)
        acc[j] = wmma_bf16(a, load_B_g(rowc[j] + kb), acc[j]);
    }
#pragma unroll
    for (int j = 0; j < 4; ++j) {
      float bias = b1[e * H2_ + ncol[j]];
#pragma unroll
      for (int r = 0; r < 8; ++r) {
        float v = gelu_exact(acc[j][r] + bias);     // D row = r + 8*hf
        ACT[(r + 8 * hf) * ASTRIDE + ncol[j]] = (__bf16)v;
      }
    }
  }
  __syncthreads();

  // ---- layer 2: out[16 x 64] = ACT[16 x 2048] @ W2^T, K split across waves ----
  {
    float* mypart = partial + wv * (16 * OUT_);
    const __bf16* Arow = ACT + lm * ASTRIDE;
    int n2[4];
    const float* roww[4];
#pragma unroll
    for (int j = 0; j < 4; ++j) {
      n2[j] = j * 16 + lm;
      roww[j] = W2 + ((size_t)e * OUT_ + n2[j]) * H2_ + 16 * hf;
    }
    v8f acc[4] = {};
#pragma unroll
    for (int ks = 0; ks < 8; ++ks) {
      int kb = (wv * 8 + ks) * 32;
      v16bf a = load_A_lds(Arow, kb, hf);
#pragma unroll
      for (int j = 0; j < 4; ++j)
        acc[j] = wmma_bf16(a, load_B_g(roww[j] + kb), acc[j]);
    }
#pragma unroll
    for (int j = 0; j < 4; ++j)
#pragma unroll
      for (int r = 0; r < 8; ++r)
        mypart[(r + 8 * hf) * OUT_ + n2[j]] = acc[j][r];
  }
  __syncthreads();

  // deterministic 8-way reduction, +b2, *routing weight, scatter
  for (int i = tid; i < 16 * OUT_; i += 256) {
    int m = i >> 6, n = i & 63;
    int t = toks[m];
    if (t < 0) continue;
    float s = 0.f;
#pragma unroll
    for (int w = 0; w < 8; ++w) s += partial[w * (16 * OUT_) + i];
    s += b2[e * OUT_ + n];
    out[(size_t)t * (E_ * OUT_) + e * OUT_ + n] = s * ew[(size_t)t * E_ + e];
  }
}

// ---------------------------------------------------------------------------
extern "C" void kernel_launch(void* const* d_in, const int* in_sizes, int n_in,
                              void* d_out, int out_size, void* d_ws, size_t ws_size,
                              hipStream_t stream) {
  const float* h      = (const float*)d_in[0];
  const float* ce     = (const float*)d_in[1];
  const float* gumbel = (const float*)d_in[2];
  const float* anchor = (const float*)d_in[3];
  const float* fi     = (const float*)d_in[4];
  const float* temp   = (const float*)d_in[5];
  const float* W1h    = (const float*)d_in[6];
  const float* W1c    = (const float*)d_in[7];
  const float* b1     = (const float*)d_in[8];
  const float* W2     = (const float*)d_in[9];
  const float* b2     = (const float*)d_in[10];
  float* out = (float*)d_out;

  char* ws = (char*)d_ws;
  float* na      = (float*)ws; ws += E_ * DC_ * sizeof(float);
  float* imp     = (float*)ws; ws += E_ * DH_ * sizeof(float);
  float* ent_imp = (float*)ws; ws += 64;
  float* ew      = (float*)ws; ws += (size_t)NTOK * E_ * sizeof(float);
  float* counts  = (float*)ws; ws += TT * E_ * sizeof(float);
  int* cnt       = (int*)ws;   ws += 64;
  int* toklist   = (int*)ws;   ws += (size_t)E_ * NTOK * sizeof(int);

  k_zero<<<4096, 256, 0, stream>>>(out, out_size, cnt);
  k_norm_anchor<<<E_, 256, 0, stream>>>(anchor, na);
  k_imp<<<E_, 256, 0, stream>>>(fi, temp, imp, ent_imp);
  k_router<<<NTOK, 256, 0, stream>>>(ce, na, gumbel, ew, cnt, toklist);
  k_counts<<<(TT * E_ + 255) / 256, 256, 0, stream>>>(ew, counts);
  k_loss<<<1, 256, 0, stream>>>(counts, ent_imp, out + (size_t)NTOK * E_ * OUT_);

  dim3 grid(NTOK / 16, E_);  // tiles x experts; blocks past cnt[e] exit early
  k_moe<<<grid, 256, 0, stream>>>(h, ce, imp, W1h, W1c, b1, W2, b2, ew, cnt,
                                  toklist, out);
}
